// GCNEncoder_18734647345386
// MI455X (gfx1250) — matrix operand order repaired and checked by reading, not compiled, not run on hardware
//
#include <hip/hip_runtime.h>
#include <hip/hip_bf16.h>
#include <stdint.h>

// GCN encoder for MI455X (gfx1250, wave32).
// Dense GEMMs run on v_wmma_f32_16x16x32_bf16 (bf16 in, f32 accumulate);
// graph propagation (the HBM-bound part) uses coalesced gathers + f32 atomics.

#define IN_CH  256
#define HID_CH 512
#define OUT_CH 256

typedef __attribute__((ext_vector_type(8)))  float  v8f;
typedef __attribute__((ext_vector_type(16))) __bf16 v16bf;

union FragBF {
  v16bf bf;
  uint4 q[2];
};

__device__ __forceinline__ unsigned short f32_to_bf16_rne(float f) {
  unsigned int u = __float_as_uint(f);
  u += 0x7FFFu + ((u >> 16) & 1u);   // round-to-nearest-even
  return (unsigned short)(u >> 16);
}

// ---------------------------------------------------------------- degree/norm
__global__ void k_fill1(float* __restrict__ d, int n) {
  int i = blockIdx.x * blockDim.x + threadIdx.x;
  if (i < n) d[i] = 1.0f;  // self-loop weight
}

__global__ void k_deg_acc(float* __restrict__ deg, const long long* __restrict__ dst,
                          const float* __restrict__ w, long long E) {
  long long e = (long long)blockIdx.x * blockDim.x + threadIdx.x;
  if (e < E) atomicAdd(&deg[(int)dst[e]], w[e]);
}

__global__ void k_rsqrt(float* __restrict__ d, int n) {
  int i = blockIdx.x * blockDim.x + threadIdx.x;
  if (i < n) {
    float v = d[i];
    d[i] = (v > 0.0f) ? rsqrtf(v) : 0.0f;
  }
}

__global__ void k_edge_norm(float* __restrict__ norm, const long long* __restrict__ src,
                            const long long* __restrict__ dst, const float* __restrict__ w,
                            const float* __restrict__ dinv, long long E) {
  long long e = (long long)blockIdx.x * blockDim.x + threadIdx.x;
  if (e < E) norm[e] = dinv[(int)src[e]] * w[e] * dinv[(int)dst[e]];
}

// ---------------------------------------------------------------- conversions
__global__ void k_cvt_bf16(const float* __restrict__ in, unsigned short* __restrict__ out,
                           long long n, int do_relu) {
  long long i = (long long)blockIdx.x * blockDim.x + threadIdx.x;
  if (i >= n) return;
  float v = in[i];
  if (do_relu) v = fmaxf(v, 0.0f);
  out[i] = f32_to_bf16_rne(v);
}

// W[K][Cout] (row-major, fp32) -> Wt[Cout][K] (bf16)
__global__ void k_transpose_cvt(const float* __restrict__ W, unsigned short* __restrict__ Wt,
                                int K, int Cout) {
  long long t = (long long)blockIdx.x * blockDim.x + threadIdx.x;
  if (t >= (long long)K * Cout) return;
  int o = (int)(t / K);
  int k = (int)(t % K);
  Wt[t] = f32_to_bf16_rne(W[(size_t)k * Cout + o]);
}

// ---------------------------------------------------------------- WMMA GEMM
// C[M][Ncols] (f32) = A[M][K] (bf16) @ Bt[Ncols][K]^T (bf16)
// One wave -> 16x64 output tile (4x v_wmma_f32_16x16x32_bf16 per K-step).
// Fragment layouts used here (A/B share contiguous k = 16h + s; C/D per CDNA5 ISA 7.12.2):
//   A 16x32: lane L holds row L&15; lanes<16 K=0..15, lanes>=16 K=16..31.
//   B 32x16: lane L holds col L&15; lanes<16 K=0..15, lanes>=16 K=16..31.
//   C/D:     VGPR r -> row (lane<16 ? r : 8+r), col lane&15.
__global__ __launch_bounds__(256)
void k_gemm_bf16_wmma(const unsigned short* __restrict__ A,
                      const unsigned short* __restrict__ Bt,
                      float* __restrict__ C, int M, int K, int Ncols) {
  const int lane = threadIdx.x & 31;
  const int wid  = (blockIdx.x << 3) + (threadIdx.x >> 5);
  const int nt64 = Ncols >> 6;
  const int mtile = wid / nt64;
  if (mtile >= (M >> 4)) return;
  const int ntile = wid % nt64;
  const int mbase = mtile << 4;
  const int nbase = ntile << 6;

  const int halfhi = (lane >> 4) & 1;   // 0: lanes 0-15, 1: lanes 16-31
  const int rowA  = mbase + (lane & 15);
  const int koffA = halfhi << 4;        // 0 or 16 (A K half)
  const int koffB = halfhi << 4;        // 0 or 16 (B K half)
  const int colB  = lane & 15;

  const unsigned short* aRow = A  + (size_t)rowA * K;
  const unsigned short* bP0  = Bt + (size_t)(nbase +  0 + colB) * K + koffB;
  const unsigned short* bP1  = Bt + (size_t)(nbase + 16 + colB) * K + koffB;
  const unsigned short* bP2  = Bt + (size_t)(nbase + 32 + colB) * K + koffB;
  const unsigned short* bP3  = Bt + (size_t)(nbase + 48 + colB) * K + koffB;

  v8f acc0 = {}, acc1 = {}, acc2 = {}, acc3 = {};

  for (int k = 0; k < K; k += 32) {
    FragBF a, b0, b1, b2, b3;
    a.q[0]  = *(const uint4*)(aRow + k + koffA);
    a.q[1]  = *(const uint4*)(aRow + k + 8 + koffA);
    b0.q[0] = *(const uint4*)(bP0 + k);
    b0.q[1] = *(const uint4*)(bP0 + k + 8);
    b1.q[0] = *(const uint4*)(bP1 + k);
    b1.q[1] = *(const uint4*)(bP1 + k + 8);
    b2.q[0] = *(const uint4*)(bP2 + k);
    b2.q[1] = *(const uint4*)(bP2 + k + 8);
    b3.q[0] = *(const uint4*)(bP3 + k);
    b3.q[1] = *(const uint4*)(bP3 + k + 8);

    if (k + 32 < K) __builtin_prefetch(aRow + k + 32, 0, 3);  // global_prefetch_b8

    acc0 = __builtin_amdgcn_wmma_f32_16x16x32_bf16(false, a.bf, false, b0.bf,
                                                   (short)0, acc0, false, false);
    acc1 = __builtin_amdgcn_wmma_f32_16x16x32_bf16(false, a.bf, false, b1.bf,
                                                   (short)0, acc1, false, false);
    acc2 = __builtin_amdgcn_wmma_f32_16x16x32_bf16(false, a.bf, false, b2.bf,
                                                   (short)0, acc2, false, false);
    acc3 = __builtin_amdgcn_wmma_f32_16x16x32_bf16(false, a.bf, false, b3.bf,
                                                   (short)0, acc3, false, false);
  }

  const int rBase = mbase + (halfhi << 3);
  const int col   = nbase + colB;
#pragma unroll
  for (int r = 0; r < 8; ++r) {
    float* cRow = C + (size_t)(rBase + r) * Ncols + col;
    cRow[0]  = acc0[r];
    cRow[16] = acc1[r];
    cRow[32] = acc2[r];
    cRow[48] = acc3[r];
  }
}

// ---------------------------------------------------------------- propagation
// out[i][c] = bias[c] + dinv[i]^2 * hlin[i][c]   (self-loop, non-atomic init)
__global__ void k_scatter_init(const float* __restrict__ hlin, const float* __restrict__ dinv,
                               const float* __restrict__ bias, float* __restrict__ out,
                               int C, long long total) {
  long long t = (long long)blockIdx.x * blockDim.x + threadIdx.x;
  if (t >= total) return;
  int i = (int)(t / C);
  int c = (int)(t % C);
  float dv = dinv[i];
  out[t] = bias[c] + dv * dv * hlin[t];
}

// out[dst] += norm[e] * hlin[src]   (one thread per (edge, float4 chunk))
__global__ void k_scatter_edges(const float* __restrict__ hlin,
                                const long long* __restrict__ src,
                                const long long* __restrict__ dst,
                                const float* __restrict__ norm,
                                float* __restrict__ out, int C, long long E) {
  long long t = (long long)blockIdx.x * blockDim.x + threadIdx.x;
  int perEdge = C >> 2;
  long long e = t / perEdge;
  if (e >= E) return;
  int c4 = (int)(t % perEdge) << 2;
  int s = (int)src[e];
  int d = (int)dst[e];
  float nv = norm[e];
  const float4 h = *(const float4*)(hlin + (size_t)s * C + c4);
  float* o = out + (size_t)d * C + c4;
  atomicAdd(o + 0, nv * h.x);
  atomicAdd(o + 1, nv * h.y);
  atomicAdd(o + 2, nv * h.z);
  atomicAdd(o + 3, nv * h.w);
}

__global__ void k_softplus_inplace(float* __restrict__ o, long long n) {
  long long i = (long long)blockIdx.x * blockDim.x + threadIdx.x;
  if (i >= n) return;
  float x = o[i];
  o[i] = (x > 0.0f) ? x + log1pf(expf(-x)) : log1pf(expf(x));
}

// ---------------------------------------------------------------- launcher
static inline unsigned int gblocks(long long n, int b) {
  return (unsigned int)((n + b - 1) / b);
}

extern "C" void kernel_launch(void* const* d_in, const int* in_sizes, int n_in,
                              void* d_out, int out_size, void* d_ws, size_t ws_size,
                              hipStream_t stream) {
  (void)n_in; (void)out_size; (void)ws_size;
  const float*     x   = (const float*)d_in[0];
  const long long* ei  = (const long long*)d_in[1];   // int64 [2][E]
  const float*     ea  = (const float*)d_in[2];
  const float*     W0  = (const float*)d_in[3];
  const float*     b0  = (const float*)d_in[4];
  const float*     W1  = (const float*)d_in[5];
  const float*     b1  = (const float*)d_in[6];
  const float*     Wmu = (const float*)d_in[7];
  const float*     bmu = (const float*)d_in[8];
  const float*     Wlv = (const float*)d_in[9];
  const float*     blv = (const float*)d_in[10];

  const int       N = in_sizes[0] / IN_CH;
  const long long E = in_sizes[2];
  const long long* srcI = ei;
  const long long* dstI = ei + E;

  // ---- workspace carving (256B aligned) ----
  char*  ws  = (char*)d_ws;
  size_t off = 0;
  auto carve = [&](size_t bytes) -> void* {
    void* p = ws + off;
    off += (bytes + 255) & ~(size_t)255;
    return p;
  };
  float*          dinv = (float*)carve((size_t)N * 4);
  float*          norm = (float*)carve((size_t)E * 4);
  unsigned short* Xbf  = (unsigned short*)carve((size_t)N * IN_CH  * 2);
  unsigned short* Hbf  = (unsigned short*)carve((size_t)N * HID_CH * 2);
  float*          Hlin = (float*)carve((size_t)N * HID_CH * 4);
  float*          Hout = (float*)carve((size_t)N * HID_CH * 4);
  unsigned short* W0t  = (unsigned short*)carve((size_t)IN_CH  * HID_CH * 2);
  unsigned short* W1t  = (unsigned short*)carve((size_t)HID_CH * HID_CH * 2);
  unsigned short* Wmut = (unsigned short*)carve((size_t)HID_CH * OUT_CH * 2);
  unsigned short* Wlvt = (unsigned short*)carve((size_t)HID_CH * OUT_CH * 2);

  const int B = 256;
  // ---- GCN normalization ----
  k_fill1    <<<gblocks(N, B), B, 0, stream>>>(dinv, N);
  k_deg_acc  <<<gblocks(E, B), B, 0, stream>>>(dinv, dstI, ea, E);
  k_rsqrt    <<<gblocks(N, B), B, 0, stream>>>(dinv, N);
  k_edge_norm<<<gblocks(E, B), B, 0, stream>>>(norm, srcI, dstI, ea, dinv, E);

  // ---- one-time conversions ----
  k_cvt_bf16<<<gblocks((long long)N * IN_CH, B), B, 0, stream>>>(x, Xbf, (long long)N * IN_CH, 0);
  k_transpose_cvt<<<gblocks((long long)IN_CH  * HID_CH, B), B, 0, stream>>>(W0,  W0t,  IN_CH,  HID_CH);
  k_transpose_cvt<<<gblocks((long long)HID_CH * HID_CH, B), B, 0, stream>>>(W1,  W1t,  HID_CH, HID_CH);
  k_transpose_cvt<<<gblocks((long long)HID_CH * OUT_CH, B), B, 0, stream>>>(Wmu, Wmut, HID_CH, OUT_CH);
  k_transpose_cvt<<<gblocks((long long)HID_CH * OUT_CH, B), B, 0, stream>>>(Wlv, Wlvt, HID_CH, OUT_CH);

  const long long NH = (long long)N * HID_CH;
  const long long NO = (long long)N * OUT_CH;
  unsigned int gemmH = gblocks((long long)(N / 16) * (HID_CH / 64), 8);
  unsigned int gemmO = gblocks((long long)(N / 16) * (OUT_CH / 64), 8);

  // ---- layer 0: relu(gcn(x, W0, b0)) ----
  k_gemm_bf16_wmma<<<gemmH, B, 0, stream>>>(Xbf, W0t, Hlin, N, IN_CH, HID_CH);
  k_scatter_init <<<gblocks(NH, B), B, 0, stream>>>(Hlin, dinv, b0, Hout, HID_CH, NH);
  k_scatter_edges<<<gblocks(E * (HID_CH / 4), B), B, 0, stream>>>(Hlin, srcI, dstI, norm, Hout, HID_CH, E);
  k_cvt_bf16     <<<gblocks(NH, B), B, 0, stream>>>(Hout, Hbf, NH, 1);

  // ---- layer 1: relu(gcn(h, W1, b1)) ----
  k_gemm_bf16_wmma<<<gemmH, B, 0, stream>>>(Hbf, W1t, Hlin, N, HID_CH, HID_CH);
  k_scatter_init <<<gblocks(NH, B), B, 0, stream>>>(Hlin, dinv, b1, Hout, HID_CH, NH);
  k_scatter_edges<<<gblocks(E * (HID_CH / 4), B), B, 0, stream>>>(Hlin, srcI, dstI, norm, Hout, HID_CH, E);
  k_cvt_bf16     <<<gblocks(NH, B), B, 0, stream>>>(Hout, Hbf, NH, 1);

  float* muOut = (float*)d_out;
  float* lvOut = muOut + NO;

  // ---- mu head ----
  k_gemm_bf16_wmma<<<gemmO, B, 0, stream>>>(Hbf, Wmut, Hlin, N, HID_CH, OUT_CH);
  k_scatter_init <<<gblocks(NO, B), B, 0, stream>>>(Hlin, dinv, bmu, muOut, OUT_CH, NO);
  k_scatter_edges<<<gblocks(E * (OUT_CH / 4), B), B, 0, stream>>>(Hlin, srcI, dstI, norm, muOut, OUT_CH, E);

  // ---- logvar head (softplus) ----
  k_gemm_bf16_wmma<<<gemmO, B, 0, stream>>>(Hbf, Wlvt, Hlin, N, HID_CH, OUT_CH);
  k_scatter_init <<<gblocks(NO, B), B, 0, stream>>>(Hlin, dinv, blv, lvOut, OUT_CH, NO);
  k_scatter_edges<<<gblocks(E * (OUT_CH / 4), B), B, 0, stream>>>(Hlin, srcI, dstI, norm, lvOut, OUT_CH, E);
  k_softplus_inplace<<<gblocks(NO, B), B, 0, stream>>>(lvOut, NO);
}
